// LinearAttention_3951369912721
// MI455X (gfx1250) — matrix operand order repaired and checked by reading, not compiled, not run on hardware
//
#include <hip/hip_runtime.h>
#include <hip/hip_bf16.h>
#include <stdint.h>

typedef __bf16 bf16;
typedef __attribute__((ext_vector_type(16))) __bf16 v16bf;
typedef __attribute__((ext_vector_type(8)))  float  v8f;
typedef __attribute__((ext_vector_type(4)))  int    v4i;

#define BATCH    4
#define SEQ      4096
#define DIMC     1024
#define HEADS    16
#define DH       64
#define TOKENS   (BATCH * SEQ)       // 16384
#define QKV_O    (3 * HEADS * DH)    // 3072
#define BH       (BATCH * HEADS)     // 64
#define ATTN_SCALE 8.0f
#define EPS 1e-12f

#if __has_builtin(__builtin_amdgcn_global_load_async_to_lds_b128)
#define HAS_ASYNC 1
#else
#define HAS_ASYNC 0
#endif

static __device__ __forceinline__ v8f wmma_bf16(v16bf a, v16bf b, v8f c) {
  return __builtin_amdgcn_wmma_f32_16x16x32_bf16(false, a, false, b, (short)0, c, false, false);
}

// ---- WMMA fragment loaders (ISA 7.12.2 layouts, wave32) ----
// A 16x32 bf16 from row-major [16][ld] at column offset k0.
static __device__ __forceinline__ v16bf frag_a(const bf16* p0, int ld, int k0) {
  int lane = threadIdx.x & 31;
  const bf16* p = p0 + (size_t)(lane & 15) * ld + k0 + ((lane >> 4) << 3);
  v16bf f;
#pragma unroll
  for (int j = 0; j < 8; ++j) { f[j] = p[j]; f[j + 8] = p[16 + j]; }
  return f;
}
// B 32x16 bf16: stored as [n][ld] (K contiguous per output column).
static __device__ __forceinline__ v16bf frag_b(const bf16* p0, int ld, int k0) {
  int lane = threadIdx.x & 31;
  const bf16* p = p0 + (size_t)(lane & 15) * ld + k0 + ((lane >> 4) << 3);
  v16bf f;
#pragma unroll
  for (int j = 0; j < 8; ++j) { f[j] = p[j]; f[j + 8] = p[16 + j]; }
  return f;
}

static __device__ __forceinline__ float block_reduce_sum_256(float v, float* red8) {
  int lane = threadIdx.x & 31, w = threadIdx.x >> 5;
#pragma unroll
  for (int off = 16; off; off >>= 1) v += __shfl_xor(v, off, 32);
  if (lane == 0) red8[w] = v;
  __syncthreads();
  float s = 0.f;
#pragma unroll
  for (int j = 0; j < 8; ++j) s += red8[j];
  return s;
}

// ---------------- 1) RMSNorm: xn = x/max(||x||,eps) * 32 * gamma  (bf16 out)
__global__ __launch_bounds__(256) void rmsnorm_kernel(const float* __restrict__ x,
                                                      const float* __restrict__ gamma,
                                                      bf16* __restrict__ xn) {
  __shared__ float red[8];
  int row = blockIdx.x;                       // token
  const float* src = x + (size_t)row * DIMC;
  float s = 0.f;
  for (int i = threadIdx.x; i < DIMC; i += 256) { float v = src[i]; s += v * v; }
  float tot = block_reduce_sum_256(s, red);
  float scale = 32.0f / fmaxf(sqrtf(tot), EPS);   // sqrt(1024)=32
  bf16* dst = xn + (size_t)row * DIMC;
  for (int i = threadIdx.x; i < DIMC; i += 256) dst[i] = (bf16)(src[i] * scale * gamma[i]);
}

// ---------------- 2) weights -> bf16
__global__ __launch_bounds__(256) void cvt_w_kernel(const float* __restrict__ wqkv,
                                                    const float* __restrict__ wout,
                                                    bf16* __restrict__ wqkv_b,
                                                    bf16* __restrict__ wout_b) {
  int i = blockIdx.x * 256 + threadIdx.x;
  const int nq = QKV_O * DIMC;                // 3145728
  const int no = DIMC * DIMC;                 // 1048576
  if (i < nq) wqkv_b[i] = (bf16)wqkv[i];
  else if (i < nq + no) wout_b[i - nq] = (bf16)wout[i - nq];
}

// ---------------- GEMM staging: one 128x32 bf16 slab of A and of B ----------------
#if HAS_ASYNC
typedef __attribute__((address_space(1))) v4i* gv4i_p;
typedef __attribute__((address_space(3))) v4i* lv4i_p;
static __device__ __forceinline__ void async_copy16(const bf16* g, bf16* l) {
  __builtin_amdgcn_global_load_async_to_lds_b128((gv4i_p)(g), (lv4i_p)(l), 0, 0);
}
#endif

static __device__ __forceinline__ void stage_tiles(const bf16* __restrict__ A,
                                                   const bf16* __restrict__ Bw,
                                                   int lda, int ldb, int m0, int n0, int k0,
                                                   bf16* ldsA, bf16* ldsB) {
  const int t = threadIdx.x;
  const int srow = t >> 1;
  const int scol = (t & 1) * 16;
  const bf16* ga = A + (size_t)(m0 + srow) * lda + k0 + scol;
  const bf16* gb = Bw + (size_t)(n0 + srow) * ldb + k0 + scol;
  bf16* la = ldsA + srow * 32 + scol;
  bf16* lb = ldsB + srow * 32 + scol;
#if HAS_ASYNC
  async_copy16(ga, la);
  async_copy16(ga + 8, la + 8);
  async_copy16(gb, lb);
  async_copy16(gb + 8, lb + 8);
#else
  ((uint4*)la)[0] = ((const uint4*)ga)[0];
  ((uint4*)la)[1] = ((const uint4*)ga)[1];
  ((uint4*)lb)[0] = ((const uint4*)gb)[0];
  ((uint4*)lb)[1] = ((const uint4*)gb)[1];
#endif
}

// one K=32 compute step on staged tiles; B-fragment loads pipelined ahead of WMMA
static __device__ __forceinline__ void compute_step(const bf16* ldsA, const bf16* ldsB,
                                                    v8f acc[8]) {
  const int w = threadIdx.x >> 5;
  v16bf fa = frag_a(ldsA + w * 16 * 32, 32, 0);
  v16bf fb = frag_b(ldsB, 32, 0);
#pragma unroll
  for (int nt = 0; nt < 7; ++nt) {
    v16bf fbn = frag_b(ldsB + (nt + 1) * 16 * 32, 32, 0);
    acc[nt] = wmma_bf16(fa, fb, acc[nt]);
    fb = fbn;
  }
  acc[7] = wmma_bf16(fa, fb, acc[7]);
}

// 128x128 block, K in steps of 32; double-buffered async staging when available
static __device__ __forceinline__ void gemm_mainloop(const bf16* __restrict__ A,
                                                     const bf16* __restrict__ Bw,
                                                     int lda, int ldb, int K,
                                                     int m0, int n0,
                                                     bf16* ldsA, bf16* ldsB, v8f acc[8]) {
#if HAS_ASYNC
  const int NK = K / 32;
  stage_tiles(A, Bw, lda, ldb, m0, n0, 0, ldsA, ldsB);
  for (int ks = 0; ks < NK; ++ks) {
    bf16* cA = ldsA + (ks & 1) * (128 * 32);
    bf16* cB = ldsB + (ks & 1) * (128 * 32);
    if (ks + 1 < NK) {
      bf16* nA = ldsA + ((ks + 1) & 1) * (128 * 32);
      bf16* nB = ldsB + ((ks + 1) & 1) * (128 * 32);
      stage_tiles(A, Bw, lda, ldb, m0, n0, (ks + 1) * 32, nA, nB);
      asm volatile("s_wait_asynccnt 0x4" ::: "memory");   // previous slab landed in LDS
    } else {
      asm volatile("s_wait_asynccnt 0x0" ::: "memory");
    }
    __syncthreads();
    compute_step(cA, cB, acc);
    __syncthreads();
  }
#else
  for (int k0 = 0; k0 < K; k0 += 32) {
    stage_tiles(A, Bw, lda, ldb, m0, n0, k0, ldsA, ldsB);
    __syncthreads();
    compute_step(ldsA, ldsB, acc);
    __syncthreads();
  }
#endif
}

// ---------------- 3) QKV GEMM: (16384x1024)x(3072x1024)^T -> q/k/v bf16 [(b h d)][n]
__global__ __launch_bounds__(256) void gemm_qkv_kernel(const bf16* __restrict__ xn,
                                                       const bf16* __restrict__ wqkv_b,
                                                       bf16* __restrict__ qt,
                                                       bf16* __restrict__ kt,
                                                       bf16* __restrict__ vt) {
  __shared__ bf16 ldsA[2 * 128 * 32];
  __shared__ bf16 ldsB[2 * 128 * 32];
  int m0 = blockIdx.x * 128, n0 = blockIdx.y * 128;
  v8f acc[8];
#pragma unroll
  for (int i = 0; i < 8; ++i) acc[i] = (v8f){0.f, 0.f, 0.f, 0.f, 0.f, 0.f, 0.f, 0.f};
  gemm_mainloop(xn, wqkv_b, DIMC, DIMC, DIMC, m0, n0, ldsA, ldsB, acc);

  int lane = threadIdx.x & 31, w = threadIdx.x >> 5;
  int nloc = lane & 15, mofs = (lane >> 4) * 8;
  int tok0 = m0 + w * 16 + mofs;
  int b = tok0 >> 12;                 // /SEQ
  int ncol = tok0 & (SEQ - 1);
#pragma unroll
  for (int nt = 0; nt < 8; ++nt) {
    int o = n0 + nt * 16 + nloc;
    int sel = o >> 10;                // 0=q,1=k,2=v
    int rem = o & 1023;               // h*64 + d
    int rowdst = b * 1024 + rem;      // (b*H+h)*DH + d
    bf16* base = (sel == 0) ? qt : ((sel == 1) ? kt : vt);
    union { uint4 u; bf16 h8[8]; } tmp;
#pragma unroll
    for (int r = 0; r < 8; ++r) tmp.h8[r] = (bf16)acc[nt][r];
    *(uint4*)(base + (size_t)rowdst * SEQ + ncol) = tmp.u;
  }
}

// ---------------- 4) per-row inv L2 norm over n (q rows get exp(temperature[h]))
__global__ __launch_bounds__(256) void rownorm_kernel(const bf16* __restrict__ qt,
                                                      const bf16* __restrict__ kt,
                                                      const float* __restrict__ temperature,
                                                      float* __restrict__ scale_q,
                                                      float* __restrict__ scale_k) {
  __shared__ float red[8];
  int row = blockIdx.x;               // 0..8191 : first 4096 q, rest k
  int r = row & 4095;
  bool isQ = row < 4096;
  const bf16* src = (isQ ? qt : kt) + (size_t)r * SEQ;
  float s = 0.f;
  for (int i = threadIdx.x; i < SEQ; i += 256) { float v = (float)src[i]; s += v * v; }
  float tot = block_reduce_sum_256(s, red);
  if (threadIdx.x == 0) {
    float sc = 1.0f / fmaxf(sqrtf(tot), EPS);
    if (isQ) sc *= __expf(temperature[(r >> 6) & (HEADS - 1)]);
    (isQ ? scale_q : scale_k)[r] = sc;
  }
}

// ---------------- 5) sim = 8 * scale_q*scale_k * (q k^T over n) ; per (b,h): 64x64, K=4096
__global__ __launch_bounds__(256) void sim_kernel(const bf16* __restrict__ qt,
                                                  const bf16* __restrict__ kt,
                                                  const float* __restrict__ scale_q,
                                                  const float* __restrict__ scale_k,
                                                  float* __restrict__ sim) {
  int bh = blockIdx.x;
  int lane = threadIdx.x & 31, w = threadIdx.x >> 5;
  int tm = w >> 1;                    // 0..3: d strip
  int tn0 = (w & 1) * 2;              // tiles tn0, tn0+1: e strips
  const bf16* qbase = qt + (size_t)(bh * 64 + tm * 16) * SEQ;
  const bf16* kb0 = kt + (size_t)(bh * 64 + tn0 * 16) * SEQ;
  const bf16* kb1 = kt + (size_t)(bh * 64 + (tn0 + 1) * 16) * SEQ;
  v8f acc0 = (v8f){0.f,0.f,0.f,0.f,0.f,0.f,0.f,0.f};
  v8f acc1 = acc0;
  for (int k0 = 0; k0 < SEQ; k0 += 32) {
    v16bf fa  = frag_a(qbase, SEQ, k0);
    v16bf fb0 = frag_b(kb0, SEQ, k0);
    v16bf fb1 = frag_b(kb1, SEQ, k0);
    acc0 = wmma_bf16(fa, fb0, acc0);
    acc1 = wmma_bf16(fa, fb1, acc1);
  }
  int nloc = lane & 15, mofs = (lane >> 4) * 8;
  int d0 = tm * 16 + mofs;
#pragma unroll
  for (int p = 0; p < 2; ++p) {
    int e = (tn0 + p) * 16 + nloc;
    float sk = scale_k[bh * 64 + e] * ATTN_SCALE;
    v8f a = p ? acc1 : acc0;
#pragma unroll
    for (int r = 0; r < 8; ++r)
      sim[(size_t)bh * 4096 + (d0 + r) * 64 + e] = a[r] * scale_q[bh * 64 + d0 + r] * sk;
  }
}

// ---------------- 6) softmax over e (rows of 64), f32 -> bf16 attn
__global__ __launch_bounds__(256) void softmax_kernel(const float* __restrict__ sim,
                                                      bf16* __restrict__ attn) {
  int bh = blockIdx.x;
  int lane = threadIdx.x & 31, w = threadIdx.x >> 5;
  for (int r = w; r < 64; r += 8) {
    const float* row = sim + (size_t)bh * 4096 + r * 64;
    float a = row[lane], b = row[lane + 32];
    float m = fmaxf(a, b);
#pragma unroll
    for (int off = 16; off; off >>= 1) m = fmaxf(m, __shfl_xor(m, off, 32));
    float ea = __expf(a - m), eb = __expf(b - m);
    float s = ea + eb;
#pragma unroll
    for (int off = 16; off; off >>= 1) s += __shfl_xor(s, off, 32);
    float inv = 1.0f / s;
    bf16* orow = attn + (size_t)bh * 4096 + r * 64;
    orow[lane] = (bf16)(ea * inv);
    orow[lane + 32] = (bf16)(eb * inv);
  }
}

// ---------------- 7) out = attn @ v : per (b,h) M=64(d), N=4096(n), K=64(e); v transposed in LDS
__global__ __launch_bounds__(256) void av_kernel(const bf16* __restrict__ attn,
                                                 const bf16* __restrict__ vt,
                                                 bf16* __restrict__ out_bnc) {
  __shared__ bf16 ldsB[256 * 72];     // [n_local][72] (64 e used, padded)
  int bh = blockIdx.x;
  int b = bh >> 4, h = bh & 15;
  int n0 = blockIdx.y * 256;
  int t = threadIdx.x;
  // stage v^T: ldsB[n][e] = v[e][n0+n]
  for (int e = 0; e < 64; ++e)
    ldsB[t * 72 + e] = vt[(size_t)(bh * 64 + e) * SEQ + n0 + t];
  __syncthreads();

  int lane = t & 31, w = t >> 5;
  int tm = w >> 1;                    // d strip 0..3
  int tnb = (w & 1) * 8;              // 8 n-tiles per wave
  const bf16* arow = attn + (size_t)bh * 4096 + tm * 16 * 64;
  v8f acc[8];
#pragma unroll
  for (int i = 0; i < 8; ++i) acc[i] = (v8f){0.f,0.f,0.f,0.f,0.f,0.f,0.f,0.f};
#pragma unroll
  for (int k0 = 0; k0 < 64; k0 += 32) {
    v16bf fa = frag_a(arow, 64, k0);
#pragma unroll
    for (int i = 0; i < 8; ++i) {
      v16bf fb = frag_b(ldsB + (tnb + i) * 16 * 72, 72, k0);
      acc[i] = wmma_bf16(fa, fb, acc[i]);
    }
  }
  int nloc = lane & 15, mofs = (lane >> 4) * 8;
  int d0 = tm * 16 + mofs;
#pragma unroll
  for (int i = 0; i < 8; ++i) {
    int n = n0 + (tnb + i) * 16 + nloc;
    union { uint4 u; bf16 h8[8]; } tmp;
#pragma unroll
    for (int r = 0; r < 8; ++r) tmp.h8[r] = (bf16)acc[i][r];
    *(uint4*)(out_bnc + (size_t)(b * SEQ + n) * DIMC + h * DH + d0) = tmp.u;
  }
}

// ---------------- 8) final projection: (16384x1024) x (1024x1024)^T -> f32 out
__global__ __launch_bounds__(256) void gemm_out_kernel(const bf16* __restrict__ out_bnc,
                                                       const bf16* __restrict__ wout_b,
                                                       float* __restrict__ out) {
  __shared__ bf16 ldsA[2 * 128 * 32];
  __shared__ bf16 ldsB[2 * 128 * 32];
  int m0 = blockIdx.x * 128, n0 = blockIdx.y * 128;
  v8f acc[8];
#pragma unroll
  for (int i = 0; i < 8; ++i) acc[i] = (v8f){0.f,0.f,0.f,0.f,0.f,0.f,0.f,0.f};
  gemm_mainloop(out_bnc, wout_b, DIMC, DIMC, DIMC, m0, n0, ldsA, ldsB, acc);

  int lane = threadIdx.x & 31, w = threadIdx.x >> 5;
  int nloc = lane & 15, mofs = (lane >> 4) * 8;
  int tok0 = m0 + w * 16 + mofs;
#pragma unroll
  for (int nt = 0; nt < 8; ++nt) {
    int o = n0 + nt * 16 + nloc;
#pragma unroll
    for (int r = 0; r < 8; ++r)
      out[(size_t)(tok0 + r) * DIMC + o] = acc[nt][r];
  }
}

extern "C" void kernel_launch(void* const* d_in, const int* in_sizes, int n_in,
                              void* d_out, int out_size, void* d_ws, size_t ws_size,
                              hipStream_t stream) {
  (void)in_sizes; (void)n_in; (void)out_size; (void)ws_size;
  const float* x     = (const float*)d_in[0];
  const float* gamma = (const float*)d_in[1];
  const float* wqkv  = (const float*)d_in[2];
  const float* temp  = (const float*)d_in[3];
  const float* wout  = (const float*)d_in[4];
  float* out = (float*)d_out;

  char* p = (char*)d_ws;
  bf16* xn      = (bf16*)p;  p += (size_t)TOKENS * DIMC * 2;
  bf16* wqkv_b  = (bf16*)p;  p += (size_t)QKV_O * DIMC * 2;
  bf16* wout_b  = (bf16*)p;  p += (size_t)DIMC * DIMC * 2;
  bf16* qt      = (bf16*)p;  p += (size_t)BH * DH * SEQ * 2;
  bf16* kt      = (bf16*)p;  p += (size_t)BH * DH * SEQ * 2;
  bf16* vt      = (bf16*)p;  p += (size_t)BH * DH * SEQ * 2;
  float* sc_q   = (float*)p; p += (size_t)BH * DH * 4;
  float* sc_k   = (float*)p; p += (size_t)BH * DH * 4;
  float* sim    = (float*)p; p += (size_t)BH * DH * DH * 4;
  bf16* attn    = (bf16*)p;  p += (size_t)BH * DH * DH * 2;
  bf16* out_bnc = (bf16*)p;  p += (size_t)TOKENS * DIMC * 2;

  rmsnorm_kernel<<<TOKENS, 256, 0, stream>>>(x, gamma, xn);
  cvt_w_kernel<<<(QKV_O * DIMC + DIMC * DIMC) / 256, 256, 0, stream>>>(wqkv, wout, wqkv_b, wout_b);
  gemm_qkv_kernel<<<dim3(TOKENS / 128, QKV_O / 128), 256, 0, stream>>>(xn, wqkv_b, qt, kt, vt);
  rownorm_kernel<<<2 * BH * DH, 256, 0, stream>>>(qt, kt, temp, sc_q, sc_k);
  sim_kernel<<<BH, 256, 0, stream>>>(qt, kt, sc_q, sc_k, sim);
  softmax_kernel<<<BH, 256, 0, stream>>>(sim, attn);
  av_kernel<<<dim3(BH, SEQ / 256), 256, 0, stream>>>(attn, vt, out_bnc);
  gemm_out_kernel<<<dim3(TOKENS / 128, DIMC / 128), 256, 0, stream>>>(out_bnc, wout_b, out);
}
